// MPNN_30769145709033
// MI455X (gfx1250) — matrix operand order repaired and checked by reading, not compiled, not run on hardware
//
#include <hip/hip_runtime.h>
#include <math.h>

typedef __attribute__((ext_vector_type(16))) _Float16 v16h;
typedef __attribute__((ext_vector_type(8)))  float    v8f;

#define N_NODES  200000
#define N_EDGES  400000
#define N_GRAPHS 10000
#define NODE_DIM 12
#define MD       32
#define HID      512
#define STEPS    4
#define N_TILES  (N_NODES / 16)   // 12500

__device__ __forceinline__ float sigmoidf_(float x) { return 1.f / (1.f + __expf(-x)); }

// ---- WMMA operand packing (wave32, 16x16x32 f16 -> f32) ------------------
// A (16x32 f16): lane L -> M = L%16; half = L/16; VGPR v holds K pair:
//   kk = (v&3)*2 + half*8 + (v>>2)*16
__device__ __forceinline__ v16h load_A_f16(const float* __restrict__ base, int nb, int lane) {
    const int m = lane & 15, half = lane >> 4;
    const float* row = base + (size_t)(nb + m) * MD;
    v16h a;
#pragma unroll
    for (int v = 0; v < 8; ++v) {
        const int kk = ((v & 3) * 2) + half * 8 + ((v >> 2) * 16);
        const float2 p = *(const float2*)(row + kk);
        a[2 * v]     = (_Float16)p.x;
        a[2 * v + 1] = (_Float16)p.y;
    }
    return a;
}

// B (32x16 f16): B[k][n] = W[obase+n][k], W row-major [rows, MD].
// lane L -> n = L%16; half = L/16; VGPR v holds K pair k = (v&3)*2 + half*8 + (v>>2)*16.
__device__ __forceinline__ v16h load_B_f16(const float* __restrict__ W, int obase, int lane) {
    const int n = lane & 15, half = lane >> 4;
    const float* row = W + (size_t)(obase + n) * MD;
    v16h b;
#pragma unroll
    for (int v = 0; v < 8; ++v) {
        const int k = ((v & 3) * 2) + half * 8 + ((v >> 2) * 16);
        const float2 p = *(const float2*)(row + k);
        b[2 * v]     = (_Float16)p.x;
        b[2 * v + 1] = (_Float16)p.y;
    }
    return b;
}

__device__ __forceinline__ v8f bcast8(float v) {
    v8f c;
#pragma unroll
    for (int r = 0; r < 8; ++r) c[r] = v;
    return c;
}

// ---- h0 = x @ node_w^T + node_b -----------------------------------------
__global__ void __launch_bounds__(256)
k_node_lin(const float* __restrict__ x, const float* __restrict__ nw,
           const float* __restrict__ nbias, float* __restrict__ h) {
    const int t = blockIdx.x * blockDim.x + threadIdx.x;
    if (t >= N_NODES * MD) return;
    const int n = t >> 5, o = t & 31;
    const float* xr = x + (size_t)n * NODE_DIM;
    const float* wr = nw + (size_t)o * NODE_DIM;
    float s = nbias[o];
#pragma unroll
    for (int k = 0; k < NODE_DIM; ++k) s += xr[k] * wr[k];
    h[t] = s;
}

// ---- edge messages: m[dst] += reshape(edge_w@ea+edge_b, 32x32) @ h[src] --
#define EDGES_PER_WAVE 4
__global__ void __launch_bounds__(256)
k_msg(const float* __restrict__ h, const float* __restrict__ ea,
      const int* __restrict__ src, const int* __restrict__ dst,
      const float* __restrict__ edge_w, const float* __restrict__ edge_b,
      float* __restrict__ m) {
    __shared__ float4 s_ew[MD * MD];
    __shared__ float  s_eb[MD * MD];
    for (int i = threadIdx.x; i < MD * MD; i += blockDim.x) {
        s_ew[i] = ((const float4*)edge_w)[i];   // edge_w row i = 4 floats
        s_eb[i] = edge_b[i];
    }
    __syncthreads();

    const int lane  = threadIdx.x & 31;
    const int wid   = threadIdx.x >> 5;
    const int ebase = (blockIdx.x * 8 + wid) * EDGES_PER_WAVE;

    for (int t = 0; t < EDGES_PER_WAVE; ++t) {
        const int e = ebase + t;
        if (e >= N_EDGES) return;
        const int s = src[e], d = dst[e];
        const float4 a = ((const float4*)ea)[e];
        const float hsv = h[(size_t)s * MD + lane];   // lane's element of h_src
        const float4* ewr = s_ew + lane * MD;         // rows for output channel = lane
        const float*  ebr = s_eb + lane * MD;
        float msg = 0.f;
#pragma unroll 8
        for (int j = 0; j < MD; ++j) {
            const float4 w4 = ewr[j];
            const float w = ebr[j] + w4.x * a.x + w4.y * a.y + w4.z * a.z + w4.w * a.w;
            msg += w * __shfl(hsv, j, 32);
        }
        atomicAdd(&m[(size_t)d * MD + lane], msg);    // L2-resident scatter
    }
}

// ---- GRU cell via WMMA: one wave per 16-node tile ------------------------
// Gate blocks in w_ih/w_hh (96 rows): r = rows [0,32), z = [32,64), n = [64,96).
// For channel half bp (0/1): blocks {bp, 2+bp, 4+bp}. r/z sums chain both
// WMMAs through the accumulator; n keeps gi_n / gh_n separate for r*gh_n.
__global__ void __launch_bounds__(256)
k_gru(const float* __restrict__ m, float* __restrict__ h,
      const float* __restrict__ w_ih, const float* __restrict__ w_hh,
      const float* __restrict__ b_ih, const float* __restrict__ b_hh) {
    const int lane = threadIdx.x & 31;
    const int wid  = threadIdx.x >> 5;
    const int tile = blockIdx.x * 8 + wid;
    if (tile >= N_TILES) return;
    const int nb = tile * 16;
    const int ncol = lane & 15, half = lane >> 4;

    const v16h am = load_A_f16(m, nb, lane);
    const v16h ah = load_A_f16(h, nb, lane);

#pragma unroll
    for (int bp = 0; bp < 2; ++bp) {            // channel halves [0,16) and [16,32)
        const int obR = bp * 16;                // r-gate block
        const int obZ = 32 + bp * 16;           // z-gate block
        const int obN = 64 + bp * 16;           // n-gate block

        // r gate: sigmoid(m@Wir^T + b_ir + h@Whr^T + b_hr), both WMMAs chained on C
        v8f c = bcast8(b_ih[obR + ncol] + b_hh[obR + ncol]);
        c = __builtin_amdgcn_wmma_f32_16x16x32_f16(false, am, false,
                load_B_f16(w_ih, obR, lane), (short)0, c, false, false);
        c = __builtin_amdgcn_wmma_f32_16x16x32_f16(false, ah, false,
                load_B_f16(w_hh, obR, lane), (short)0, c, false, false);
        v8f rg;
#pragma unroll
        for (int r = 0; r < 8; ++r) rg[r] = sigmoidf_(c[r]);

        // z gate
        c = bcast8(b_ih[obZ + ncol] + b_hh[obZ + ncol]);
        c = __builtin_amdgcn_wmma_f32_16x16x32_f16(false, am, false,
                load_B_f16(w_ih, obZ, lane), (short)0, c, false, false);
        c = __builtin_amdgcn_wmma_f32_16x16x32_f16(false, ah, false,
                load_B_f16(w_hh, obZ, lane), (short)0, c, false, false);
        v8f zg;
#pragma unroll
        for (int r = 0; r < 8; ++r) zg[r] = sigmoidf_(c[r]);

        // n gate: tanh(gi_n + rg * gh_n)
        v8f gin = __builtin_amdgcn_wmma_f32_16x16x32_f16(false, am, false,
                load_B_f16(w_ih, obN, lane), (short)0, bcast8(b_ih[obN + ncol]), false, false);
        v8f ghn = __builtin_amdgcn_wmma_f32_16x16x32_f16(false, ah, false,
                load_B_f16(w_hh, obN, lane), (short)0, bcast8(b_hh[obN + ncol]), false, false);

        // combine + in-place h update; C/D layout: (M = r + 8*half, N = ncol)
#pragma unroll
        for (int r = 0; r < 8; ++r) {
            const float ng = tanhf(gin[r] + rg[r] * ghn[r]);
            const size_t idx = (size_t)(nb + r + 8 * half) * MD + bp * 16 + ncol;
            const float hv = h[idx];
            h[idx] = (1.f - zg[r]) * ng + zg[r] * hv;
        }
    }
}

// ---- global mean-pool (sum + count) --------------------------------------
__global__ void __launch_bounds__(256)
k_pool(const float* __restrict__ h, const int* __restrict__ batch,
       float* __restrict__ sums, float* __restrict__ counts) {
    const int t = blockIdx.x * blockDim.x + threadIdx.x;
    if (t >= N_NODES * MD) return;
    const int n = t >> 5, c = t & 31;
    const int g = batch[n];
    atomicAdd(&sums[(size_t)g * MD + c], h[t]);
    if (c == 0) atomicAdd(&counts[g], 1.0f);
}

// ---- readout MLP: one wave per graph -------------------------------------
__global__ void __launch_bounds__(256)
k_readout(const float* __restrict__ sums, const float* __restrict__ counts,
          const float* __restrict__ w1, const float* __restrict__ b1,
          const float* __restrict__ w2, const float* __restrict__ b2,
          float* __restrict__ out) {
    const int lane = threadIdx.x & 31;
    const int wid  = threadIdx.x >> 5;
    const int g = blockIdx.x * 8 + wid;
    if (g >= N_GRAPHS) return;
    float cnt = counts[g];
    cnt = cnt > 1.f ? cnt : 1.f;
    const float hgv = sums[(size_t)g * MD + lane] / cnt;
    float acc = 0.f;
#pragma unroll
    for (int u = 0; u < HID / 32; ++u) {
        const int hid = u * 32 + lane;
        const float* wr = w1 + (size_t)hid * MD;
        float s = b1[hid];
#pragma unroll
        for (int k = 0; k < MD; ++k) s += __shfl(hgv, k, 32) * wr[k];
        s = fmaxf(s, 0.f);
        acc += s * w2[hid];
    }
#pragma unroll
    for (int off = 16; off >= 1; off >>= 1) acc += __shfl_xor(acc, off, 32);
    if (lane == 0) out[g] = sigmoidf_(acc + b2[0]);
}

extern "C" void kernel_launch(void* const* d_in, const int* in_sizes, int n_in,
                              void* d_out, int out_size, void* d_ws, size_t ws_size,
                              hipStream_t stream) {
    const float* x         = (const float*)d_in[0];
    const float* edge_attr = (const float*)d_in[1];
    const int*   edge_idx  = (const int*)d_in[2];
    const int*   batch     = (const int*)d_in[3];
    const float* node_w    = (const float*)d_in[4];
    const float* node_b    = (const float*)d_in[5];
    const float* edge_w    = (const float*)d_in[6];
    const float* edge_b    = (const float*)d_in[7];
    const float* gru_w_ih  = (const float*)d_in[8];
    const float* gru_w_hh  = (const float*)d_in[9];
    const float* gru_b_ih  = (const float*)d_in[10];
    const float* gru_b_hh  = (const float*)d_in[11];
    const float* ro_w1     = (const float*)d_in[12];
    const float* ro_b1     = (const float*)d_in[13];
    const float* ro_w2     = (const float*)d_in[14];
    const float* ro_b2     = (const float*)d_in[15];
    float* out = (float*)d_out;

    const int* src = edge_idx;            // edge_index[0]
    const int* dst = edge_idx + N_EDGES;  // edge_index[1]

    // workspace layout (fp32): h[N*MD] | m[N*MD] | sums[G*MD] | counts[G]
    float* h      = (float*)d_ws;
    float* m      = h + (size_t)N_NODES * MD;
    float* sums   = m + (size_t)N_NODES * MD;
    float* counts = sums + (size_t)N_GRAPHS * MD;

    const int elem_threads = N_NODES * MD;
    const int elem_blocks  = (elem_threads + 255) / 256;

    k_node_lin<<<elem_blocks, 256, 0, stream>>>(x, node_w, node_b, h);

    const int msg_blocks = (N_EDGES + 8 * EDGES_PER_WAVE - 1) / (8 * EDGES_PER_WAVE);
    const int gru_blocks = (N_TILES + 7) / 8;
    for (int s = 0; s < STEPS; ++s) {
        hipMemsetAsync(m, 0, (size_t)N_NODES * MD * sizeof(float), stream);
        k_msg<<<msg_blocks, 256, 0, stream>>>(h, edge_attr, src, dst, edge_w, edge_b, m);
        k_gru<<<gru_blocks, 256, 0, stream>>>(m, h, gru_w_ih, gru_w_hh, gru_b_ih, gru_b_hh);
    }

    hipMemsetAsync(sums, 0, ((size_t)N_GRAPHS * MD + N_GRAPHS) * sizeof(float), stream);
    k_pool<<<elem_blocks, 256, 0, stream>>>(h, batch, sums, counts);
    k_readout<<<(N_GRAPHS + 7) / 8, 256, 0, stream>>>(sums, counts, ro_w1, ro_b1, ro_w2, ro_b2, out);
}
